// QUIK_67654324847582
// MI455X (gfx1250) — 8-bit matrix operand order repaired and checked by reading, compiled, not run on hardware
//
#include <hip/hip_runtime.h>
#include <stdint.h>

typedef __attribute__((ext_vector_type(8))) int v8i;

#define NDIM 4096
#define KDIM 4096

__device__ __forceinline__ unsigned f2ord(float f) {
  unsigned u = __float_as_uint(f);
  return (u & 0x80000000u) ? ~u : (u | 0x80000000u);
}
__device__ __forceinline__ float ord2f(unsigned u) {
  return (u & 0x80000000u) ? __uint_as_float(u & 0x7FFFFFFFu)
                           : __uint_as_float(~u);
}

__global__ void quik_init(unsigned* mm) {
  if (threadIdx.x == 0) {
    mm[0] = 0xFFFFFFFFu;
    mm[1] = 0u;
    mm[2] = 0xFFFFFFFFu;
    mm[3] = 0u;
  }
}

__global__ void quik_minmax(const float4* __restrict__ d, int n4,
                            unsigned* __restrict__ mn,
                            unsigned* __restrict__ mx) {
  __shared__ unsigned smn[256], smx[256];
  unsigned lmn = 0xFFFFFFFFu, lmx = 0u;
  for (int i = blockIdx.x * blockDim.x + threadIdx.x; i < n4;
       i += gridDim.x * blockDim.x) {
    float4 v = d[i];
    unsigned a = f2ord(v.x), b = f2ord(v.y), c = f2ord(v.z), e = f2ord(v.w);
    lmn = min(min(lmn, a), min(b, min(c, e)));
    lmx = max(max(lmx, a), max(b, max(c, e)));
  }
  smn[threadIdx.x] = lmn; smx[threadIdx.x] = lmx;
  __syncthreads();
  for (int s = 128; s > 0; s >>= 1) {
    if ((int)threadIdx.x < s) {
      smn[threadIdx.x] = min(smn[threadIdx.x], smn[threadIdx.x + s]);
      smx[threadIdx.x] = max(smx[threadIdx.x], smx[threadIdx.x + s]);
    }
    __syncthreads();
  }
  if (threadIdx.x == 0) { atomicMin(mn, smn[0]); atomicMax(mx, smx[0]); }
}

__global__ void quik_rowsum(const float* __restrict__ w,
                            float* __restrict__ rs) {
  __shared__ float sm[256];
  const float4* p = (const float4*)(w + (size_t)blockIdx.x * KDIM);
  float s = 0.f;
  for (int i = threadIdx.x; i < KDIM / 4; i += 256) {
    float4 v = p[i];
    s += (v.x + v.y) + (v.z + v.w);
  }
  sm[threadIdx.x] = s;
  __syncthreads();
  for (int t = 128; t > 0; t >>= 1) {
    if ((int)threadIdx.x < t) sm[threadIdx.x] += sm[threadIdx.x + t];
    __syncthreads();
  }
  if (threadIdx.x == 0) rs[blockIdx.x] = sm[0];
}

__global__ void quik_quant(const float4* __restrict__ d,
                           char4* __restrict__ q, int n4,
                           const unsigned* __restrict__ mm, int slot) {
  const float zero  = ord2f(mm[slot]);
  const float scale = (ord2f(mm[slot + 1]) - zero) * 0.125f;
  for (int i = blockIdx.x * blockDim.x + threadIdx.x; i < n4;
       i += gridDim.x * blockDim.x) {
    float4 v = d[i];
    float qx = fminf(fmaxf((v.x - zero) / scale - 4.0f, -4.0f), 3.0f);
    float qy = fminf(fmaxf((v.y - zero) / scale - 4.0f, -4.0f), 3.0f);
    float qz = fminf(fmaxf((v.z - zero) / scale - 4.0f, -4.0f), 3.0f);
    float qw = fminf(fmaxf((v.w - zero) / scale - 4.0f, -4.0f), 3.0f);
    char4 o;
    o.x = (signed char)(int)qx;
    o.y = (signed char)(int)qy;
    o.z = (signed char)(int)qz;
    o.w = (signed char)(int)qw;
    q[i] = o;
  }
}

__global__ void __launch_bounds__(256) quik_gemm(
    const char* __restrict__ xq, const char* __restrict__ wq,
    const float* __restrict__ bias, const float* __restrict__ rs,
    const unsigned* __restrict__ mm, float* __restrict__ out) {
  const int lane = threadIdx.x & 31;
  const int wave = threadIdx.x >> 5;
  const int m16  = lane & 15;
  const int h    = lane >> 4;

  const int mBase = blockIdx.x * 128 + wave * 16;
  const int nBase = blockIdx.y * 128;

  const char* arow  = xq + (size_t)(mBase + m16) * KDIM;
  const char* brow  = wq + (size_t)(nBase + m16) * KDIM;

  v8i acc[8];
  #pragma unroll
  for (int t = 0; t < 8; ++t) acc[t] = (v8i){0, 0, 0, 0, 0, 0, 0, 0};

  for (int kb = 0; kb < KDIM; kb += 64) {
    __builtin_prefetch(arow + kb + 64, 0, 3);
    __builtin_prefetch(brow + kb + 64, 0, 3);

    const int aoff = kb + h * 16;
    int4 alo = *(const int4*)(arow + aoff);
    int4 ahi = *(const int4*)(arow + aoff + 32);
    v8i a;
    a[0] = alo.x; a[1] = alo.y; a[2] = alo.z; a[3] = alo.w;
    a[4] = ahi.x; a[5] = ahi.y; a[6] = ahi.z; a[7] = ahi.w;

    const int boff = kb + h * 16;
    #pragma unroll
    for (int t = 0; t < 8; ++t) {
      const char* bp = brow + (size_t)t * (16 * KDIM);
      int4 lo = *(const int4*)(bp + boff);
      int4 hi = *(const int4*)(bp + boff + 32);
      v8i b;
      b[0] = lo.x; b[1] = lo.y; b[2] = lo.z; b[3] = lo.w;
      b[4] = hi.x; b[5] = hi.y; b[6] = hi.z; b[7] = hi.w;
      acc[t] = __builtin_amdgcn_wmma_i32_16x16x64_iu8(true, a, true, b,
                                                      acc[t], false, false);
    }
  }

  const float zero_act = ord2f(mm[0]);
  const float sa  = (ord2f(mm[1]) - zero_act) * 0.125f;
  const float wmn = ord2f(mm[2]);
  const float sw  = (ord2f(mm[3]) - wmn) * 0.125f;
  const float sasw = sa * sw;
  const float zp4  = zero_act + 4.0f * sa;

  const int col0 = nBase + m16;
  float bj[8];
  #pragma unroll
  for (int t = 0; t < 8; ++t) bj[t] = bias[col0 + 16 * t] * sasw;

  #pragma unroll
  for (int r = 0; r < 8; ++r) {
    const int m = mBase + r + 8 * h;
    const float shift = zp4 * rs[m];
    float* orow = out + (size_t)m * NDIM + col0;
    #pragma unroll
    for (int t = 0; t < 8; ++t) {
      float v = (float)acc[t][r] * sasw + bj[t] + shift;
      __builtin_nontemporal_store(v, orow + 16 * t);
    }
  }
}

extern "C" void kernel_launch(void* const* d_in, const int* in_sizes, int n_in,
                              void* d_out, int out_size, void* d_ws,
                              size_t ws_size, hipStream_t stream) {
  const float* x    = (const float*)d_in[0];
  const float* w    = (const float*)d_in[1];
  const float* bias = (const float*)d_in[2];
  float* out = (float*)d_out;

  unsigned* mm = (unsigned*)d_ws;
  float* rs = (float*)((char*)d_ws + 64);
  char* xq = (char*)d_ws + 64 + NDIM * 4;
  char* wq = xq + (size_t)NDIM * KDIM;

  const int n4 = NDIM * KDIM / 4;

  quik_init<<<1, 64, 0, stream>>>(mm);
  quik_minmax<<<1024, 256, 0, stream>>>((const float4*)x, n4, mm + 0, mm + 1);
  quik_minmax<<<1024, 256, 0, stream>>>((const float4*)w, n4, mm + 2, mm + 3);
  quik_rowsum<<<NDIM, 256, 0, stream>>>(w, rs);
  quik_quant<<<2048, 256, 0, stream>>>((const float4*)x, (char4*)xq, n4, mm, 0);
  quik_quant<<<2048, 256, 0, stream>>>((const float4*)w, (char4*)wq, n4, mm, 2);

  dim3 grid(NDIM / 128, NDIM / 128);
  quik_gemm<<<grid, 256, 0, stream>>>(xq, wq, bias, rs, mm, out);
}
